// TransMIL_42795054137565
// MI455X (gfx1250) — compile-verified, matrix operand order checked by reading, not run on hardware
//
#include <hip/hip_runtime.h>
#include <hip/hip_bf16.h>

// ---------------- model constants (fixed shapes from setup_inputs) ----------
constexpr int HEADS   = 8;
constexpr int DH      = 64;
constexpr int MLAND   = 256;
constexpr int PINVIT  = 6;
constexpr int NTOK    = 10000;   // Hn
constexpr int TTOK    = 10001;   // tokens incl. cls
constexpr int EIN     = 1024;
constexpr int DMODEL  = 512;
constexpr int NPAD    = 10240;   // ceil(10001/256)*256
constexpr int PADROWS = 239;     // NPAD - TTOK
constexpr int LGRP    = 40;      // NPAD / MLAND
constexpr int HGRID   = 100;     // sqrt(NTOK)
constexpr int AHID    = 256;     // attention-head hidden

typedef __attribute__((ext_vector_type(16))) _Float16 v16h;
typedef __attribute__((ext_vector_type(8)))  float    v8f;
typedef __attribute__((ext_vector_type(2)))  float    v2f;

// ---------------- WMMA f16 GEMM (fp32 in/out, f32 accumulate) ---------------
// A: MxK row-major fp32, B: KxN row-major fp32, C: MxN fp32.
// Each wave computes a 16x64 strip: A fragment converted once, reused by 4
// v_wmma_f32_16x16x32_f16. Fragment layouts per CDNA5 ISA 7.12.2.
__device__ __forceinline__ int wmma_koff(int j, int half_id) {
  return ((j < 8) ? j : (j + 8)) + (half_id << 3);
}

// ACT: 0 none, 1 relu, 2 tanh.  ACC: 0 store, 1 accumulate (+=).
template <int ACT, int ACC>
__global__ void k_wmma_gemm(const float* __restrict__ A, const float* __restrict__ B,
                            const float* __restrict__ bias, float* __restrict__ C,
                            int M, int N, int K) {
  const int wavesPerBlock = blockDim.x >> 5;
  const int wave = blockIdx.x * wavesPerBlock + (threadIdx.x >> 5);
  const int lane = threadIdx.x & 31;
  const int tilesN4 = N >> 6;               // 64-wide strips
  const int tilesM = (M + 15) >> 4;
  if (wave >= tilesM * tilesN4) return;
  const int tm = wave / tilesN4, tn4 = wave % tilesN4;
  const int half_id = lane >> 4;
  const int r = lane & 15;
  const int arow = tm * 16 + r;             // A row held by this lane
  const int bcol0 = tn4 * 64 + r;           // first B column held by this lane

  union U8 { v8f v; float f[8]; } acc[4];
#pragma unroll
  for (int tt = 0; tt < 4; ++tt)
#pragma unroll
    for (int i = 0; i < 8; ++i) acc[tt].f[i] = 0.0f;

  const float* Arow = A + (size_t)arow * K;
  const float* Bbase = B + bcol0;
  for (int k0 = 0; k0 < K; k0 += 32) {
    if (k0 + 32 < K) {
      __builtin_prefetch(Arow + k0 + 32, 0, 3);          // global_prefetch_b8
      __builtin_prefetch(Bbase + (size_t)(k0 + 32) * N, 0, 3);
    }
    union { v16h v; _Float16 h[16]; } af;
#pragma unroll
    for (int j = 0; j < 16; ++j) {
      const int kk = k0 + wmma_koff(j, half_id);
      af.h[j] = (arow < M) ? (_Float16)Arow[kk] : (_Float16)0.0f;
    }
#pragma unroll
    for (int tt = 0; tt < 4; ++tt) {
      union { v16h v; _Float16 h[16]; } bf;
#pragma unroll
      for (int j = 0; j < 16; ++j) {
        const int kk = k0 + wmma_koff(j, half_id);
        bf.h[j] = (_Float16)Bbase[(size_t)kk * N + 16 * tt];
      }
      acc[tt].v = __builtin_amdgcn_wmma_f32_16x16x32_f16(false, af.v, false, bf.v,
                                                         (short)0, acc[tt].v, false, false);
    }
  }
#pragma unroll
  for (int tt = 0; tt < 4; ++tt) {
    const int bcol = bcol0 + 16 * tt;
    const float bv = bias ? bias[bcol] : 0.0f;
#pragma unroll
    for (int rr = 0; rr < 8; ++rr) {
      const int m = tm * 16 + rr + (half_id << 3);
      if (m < M) {
        float val = acc[tt].f[rr] + bv;
        if (ACT == 1) val = fmaxf(val, 0.0f);
        if (ACT == 2) val = tanhf(val);
        const size_t idx = (size_t)m * N + bcol;
        if (ACC) C[idx] += val; else C[idx] = val;
      }
    }
  }
}

// QKV projection: C = XLN(NPADxD) @ Wqkv(Dx3D), scattered into head-major
// Q/K/V buffers of shape [HEADS][NPAD][DH]; q scaled by DH^-0.5.
__global__ void k_wmma_qkv(const float* __restrict__ A, const float* __restrict__ B,
                           float* __restrict__ Qb, float* __restrict__ Kb,
                           float* __restrict__ Vb) {
  const int M = NPAD, N = 3 * DMODEL, K = DMODEL;
  const int wavesPerBlock = blockDim.x >> 5;
  const int wave = blockIdx.x * wavesPerBlock + (threadIdx.x >> 5);
  const int lane = threadIdx.x & 31;
  const int tilesN4 = N >> 6;
  const int tilesM = M >> 4;
  if (wave >= tilesM * tilesN4) return;
  const int tm = wave / tilesN4, tn4 = wave % tilesN4;
  const int half_id = lane >> 4;
  const int r = lane & 15;
  const int arow = tm * 16 + r;
  const int bcol0 = tn4 * 64 + r;

  union U8 { v8f v; float f[8]; } acc[4];
#pragma unroll
  for (int tt = 0; tt < 4; ++tt)
#pragma unroll
    for (int i = 0; i < 8; ++i) acc[tt].f[i] = 0.0f;

  const float* Arow = A + (size_t)arow * K;
  const float* Bbase = B + bcol0;
  for (int k0 = 0; k0 < K; k0 += 32) {
    if (k0 + 32 < K) {
      __builtin_prefetch(Arow + k0 + 32, 0, 3);
      __builtin_prefetch(Bbase + (size_t)(k0 + 32) * N, 0, 3);
    }
    union { v16h v; _Float16 h[16]; } af;
#pragma unroll
    for (int j = 0; j < 16; ++j) {
      const int kk = k0 + wmma_koff(j, half_id);
      af.h[j] = (_Float16)Arow[kk];
    }
#pragma unroll
    for (int tt = 0; tt < 4; ++tt) {
      union { v16h v; _Float16 h[16]; } bf;
#pragma unroll
      for (int j = 0; j < 16; ++j) {
        const int kk = k0 + wmma_koff(j, half_id);
        bf.h[j] = (_Float16)Bbase[(size_t)kk * N + 16 * tt];
      }
      acc[tt].v = __builtin_amdgcn_wmma_f32_16x16x32_f16(false, af.v, false, bf.v,
                                                         (short)0, acc[tt].v, false, false);
    }
  }
#pragma unroll
  for (int tt = 0; tt < 4; ++tt) {
    const int bcol = bcol0 + 16 * tt;
    const int which = bcol >> 9;          // 0=q 1=k 2=v
    const int cc    = bcol & 511;
    const int head  = cc >> 6;
    const int dh    = cc & 63;
    const float scale = (which == 0) ? 0.125f : 1.0f;   // DH^-0.5
    float* dst = (which == 0) ? Qb : (which == 1) ? Kb : Vb;
#pragma unroll
    for (int rr = 0; rr < 8; ++rr) {
      const int m = tm * 16 + rr + (half_id << 3);
      dst[((size_t)head * NPAD + m) * DH + dh] = acc[tt].f[rr] * scale;
    }
  }
}

// Batched fp32 matmul on the matrix pipe: C[h] = scale*(A[h] @ B[h]) using
// V_WMMA_F32_16X16X4_F32 (full fp32 — needed for Newton-Schulz stability).
// 32-bit A 16x4 layout: lanes 0-15 hold K=0,1; lanes 16-31 hold K=2,3.
__global__ void k_bmm(const float* __restrict__ A, const float* __restrict__ B,
                      float* __restrict__ C, int Md, int Nd, int Kd, float scale) {
  const int h = blockIdx.z;
  const float* Ah = A + (size_t)h * Md * Kd;
  const float* Bh = B + (size_t)h * Kd * Nd;
  float* Ch = C + (size_t)h * Md * Nd;
  const int wavesPerBlock = blockDim.x >> 5;
  const int wave = blockIdx.x * wavesPerBlock + (threadIdx.x >> 5);
  const int lane = threadIdx.x & 31;
  const int tilesN = Nd >> 4;
  if (wave >= (Md >> 4) * tilesN) return;
  const int tm = wave / tilesN, tn = wave % tilesN;
  const int half_id = lane >> 4;
  const int r = lane & 15;
  const int arow = tm * 16 + r;
  const int bcol = tn * 16 + r;

  union { v8f v; float f[8]; } acc;
#pragma unroll
  for (int i = 0; i < 8; ++i) acc.f[i] = 0.0f;

  const float* Ap = Ah + (size_t)arow * Kd + 2 * half_id;   // K = k0 + 2*half + j
  const float* Bp = Bh + bcol + (size_t)(2 * half_id) * Nd;
  for (int k0 = 0; k0 < Kd; k0 += 4) {
    union { v2f v; float f[2]; } a, b;
    a.f[0] = Ap[k0];
    a.f[1] = Ap[k0 + 1];
    b.f[0] = Bp[(size_t)k0 * Nd];
    b.f[1] = Bp[(size_t)(k0 + 1) * Nd];
    acc.v = __builtin_amdgcn_wmma_f32_16x16x4_f32(false, a.v, false, b.v,
                                                  (short)0, acc.v, false, false);
  }
#pragma unroll
  for (int rr = 0; rr < 8; ++rr) {
    const int m = tm * 16 + rr + (half_id << 3);
    Ch[(size_t)m * Nd + bcol] = acc.f[rr] * scale;
  }
}

// ---------------- elementwise / small kernels -------------------------------
__global__ void k_set_cls(const float* __restrict__ cls, float* __restrict__ X) {
  const int t = threadIdx.x;
  if (t < DMODEL) X[t] = cls[t];
}

// LayerNorm rows of X (TTOK x D), zero-pad PADROWS rows at front -> O (NPAD x D)
__global__ void k_ln_pad(const float* __restrict__ X, const float* __restrict__ g,
                         const float* __restrict__ b, float* __restrict__ O) {
  const int row = blockIdx.x;
  const int t = threadIdx.x;     // 256
  __shared__ float red[256];
  float* out = O + (size_t)row * DMODEL;
  if (row < PADROWS) {
    for (int c = t; c < DMODEL; c += 256) out[c] = 0.0f;
    return;
  }
  const float* in = X + (size_t)(row - PADROWS) * DMODEL;
  float s = 0.0f;
  for (int c = t; c < DMODEL; c += 256) s += in[c];
  red[t] = s; __syncthreads();
  for (int o = 128; o > 0; o >>= 1) { if (t < o) red[t] += red[t + o]; __syncthreads(); }
  const float mu = red[0] / DMODEL; __syncthreads();
  float vs = 0.0f;
  for (int c = t; c < DMODEL; c += 256) { const float d = in[c] - mu; vs += d * d; }
  red[t] = vs; __syncthreads();
  for (int o = 128; o > 0; o >>= 1) { if (t < o) red[t] += red[t + o]; __syncthreads(); }
  const float rstd = rsqrtf(red[0] / DMODEL + 1e-5f); __syncthreads();
  for (int c = t; c < DMODEL; c += 256)
    out[c] = (in[c] - mu) * rstd * g[c] + b[c];
}

// Landmarks: QL[h][i][d] = mean over LGRP contiguous rows of Q[h]
__global__ void k_landmarks(const float* __restrict__ Qb, float* __restrict__ QL) {
  const int hi = blockIdx.x;            // h*MLAND + i
  const int h = hi >> 8, i = hi & 255;
  const int d = threadIdx.x;            // 64
  const float* base = Qb + ((size_t)h * NPAD + (size_t)i * LGRP) * DH + d;
  float s = 0.0f;
  for (int j = 0; j < LGRP; ++j) s += base[(size_t)j * DH];
  QL[((size_t)h * MLAND + i) * DH + d] = s * (1.0f / LGRP);
}

// a2 = softmax(ql @ kl^T) per head: block per (h,i), thread per j.
__global__ void k_a2(const float* __restrict__ QL, const float* __restrict__ KL,
                     float* __restrict__ A2) {
  const int hi = blockIdx.x; const int h = hi >> 8, i = hi & 255;
  const int j = threadIdx.x;            // 256
  __shared__ float qv[DH];
  __shared__ float red[256];
  if (j < DH) qv[j] = QL[((size_t)h * MLAND + i) * DH + j];
  __syncthreads();
  const float* kr = KL + ((size_t)h * MLAND + j) * DH;
  float s = 0.0f;
  for (int d = 0; d < DH; ++d) s += qv[d] * kr[d];
  red[j] = s; __syncthreads();
  for (int o = 128; o > 0; o >>= 1) { if (j < o) red[j] = fmaxf(red[j], red[j + o]); __syncthreads(); }
  const float mx = red[0]; __syncthreads();
  const float e = expf(s - mx);
  red[j] = e; __syncthreads();
  for (int o = 128; o > 0; o >>= 1) { if (j < o) red[j] += red[j + o]; __syncthreads(); }
  A2[((size_t)h * MLAND + i) * MLAND + j] = e / red[0];
}

__global__ void k_zero4(float* __restrict__ s) { if (threadIdx.x < 8) s[threadIdx.x] = 0.0f; }

// row/col abs-sum global max -> SCAL[0], SCAL[1] (float atomicMax via int bits, values>=0)
__global__ void k_rowcol(const float* __restrict__ A2, float* __restrict__ SCAL) {
  const int idx = blockIdx.x * blockDim.x + threadIdx.x;
  if (idx >= HEADS * MLAND) return;
  const int h = idx >> 8, i = idx & 255;
  const float* base = A2 + (size_t)h * MLAND * MLAND;
  float s = 0.0f;
  if (blockIdx.y == 0) {
    const float* rowp = base + (size_t)i * MLAND;
    for (int j = 0; j < MLAND; ++j) s += fabsf(rowp[j]);
  } else {
    for (int j = 0; j < MLAND; ++j) s += fabsf(base[(size_t)j * MLAND + i]);
  }
  atomicMax((int*)(SCAL + blockIdx.y), __float_as_int(s));
}

__global__ void k_pinv_init(const float* __restrict__ A2, const float* __restrict__ SCAL,
                            float* __restrict__ Z) {
  const size_t idx = (size_t)blockIdx.x * 256 + threadIdx.x;
  if (idx >= (size_t)HEADS * MLAND * MLAND) return;
  const int h = (int)(idx >> 16);
  const int rem = (int)(idx & 65535);
  const int i = rem >> 8, j = rem & 255;
  Z[idx] = A2[((size_t)h * MLAND + j) * MLAND + i] / (SCAL[0] * SCAL[1]);
}

// Dst = coef*I - S   (per-head 256x256)
__global__ void k_diagsub(const float* __restrict__ S, float* __restrict__ Dst, float coef) {
  const size_t idx = (size_t)blockIdx.x * 256 + threadIdx.x;
  if (idx >= (size_t)HEADS * MLAND * MLAND) return;
  const int rem = (int)(idx & 65535);
  const int i = rem >> 8, j = rem & 255;
  Dst[idx] = ((i == j) ? coef : 0.0f) - S[idx];
}

// W3[h][i] = softmax(ql_i @ K^T over NPAD keys) @ V   — streamed with full
// logit row (40KB) resident in LDS (320KB/WGP budget).
__global__ void __launch_bounds__(256) k_a3v(const float* __restrict__ QL,
                                             const float* __restrict__ Kb,
                                             const float* __restrict__ Vb,
                                             float* __restrict__ W3) {
  const int hi = blockIdx.x; const int h = hi >> 8, i = hi & 255;
  const int t = threadIdx.x;            // 256
  __shared__ float logits[NPAD];
  __shared__ float qv[DH];
  __shared__ float red[256];
  if (t < DH) qv[t] = QL[((size_t)h * MLAND + i) * DH + t];
  __syncthreads();
  float mx = -3.4e38f;
  for (int n = t; n < NPAD; n += 256) {
    const float* kr = Kb + ((size_t)h * NPAD + n) * DH;
    float s = 0.0f;
    for (int d = 0; d < DH; ++d) s += qv[d] * kr[d];
    logits[n] = s;
    mx = fmaxf(mx, s);
  }
  red[t] = mx; __syncthreads();
  for (int o = 128; o > 0; o >>= 1) { if (t < o) red[t] = fmaxf(red[t], red[t + o]); __syncthreads(); }
  mx = red[0]; __syncthreads();
  float sm = 0.0f;
  for (int n = t; n < NPAD; n += 256) { const float e = expf(logits[n] - mx); logits[n] = e; sm += e; }
  red[t] = sm; __syncthreads();
  for (int o = 128; o > 0; o >>= 1) { if (t < o) red[t] += red[t + o]; __syncthreads(); }
  const float inv = 1.0f / red[0]; __syncthreads();
  const int d = t & 63, g = t >> 6;     // 4 partial groups x 64 dims
  float acc = 0.0f;
  for (int n = g; n < NPAD; n += 4)
    acc += logits[n] * Vb[((size_t)h * NPAD + n) * DH + d];
  red[t] = acc; __syncthreads();
  if (g == 0)
    W3[((size_t)h * MLAND + i) * DH + d] =
        (red[d] + red[64 + d] + red[128 + d] + red[192 + d]) * inv;
}

// out[n, h*64+d] = softmax(q_n @ kl^T) @ WZ[h]  + depthwise 33-tap conv(V).
__global__ void __launch_bounds__(64) k_attn_out(const float* __restrict__ Qb,
                                                 const float* __restrict__ KL,
                                                 const float* __restrict__ WZ,
                                                 const float* __restrict__ Vb,
                                                 const float* __restrict__ resw,
                                                 float* __restrict__ OUT) {
  const int h = blockIdx.y;
  const int n = blockIdx.x;
  const int t = threadIdx.x;            // 64
  __shared__ float qv[DH];
  __shared__ float p[MLAND];
  __shared__ float red[64];
  qv[t] = Qb[((size_t)h * NPAD + n) * DH + t];
  __syncthreads();
  float lm = -3.4e38f;
#pragma unroll
  for (int jj = 0; jj < 4; ++jj) {
    const int j = t * 4 + jj;
    const float* kr = KL + ((size_t)h * MLAND + j) * DH;
    float s = 0.0f;
    for (int d = 0; d < DH; ++d) s += qv[d] * kr[d];
    p[j] = s; lm = fmaxf(lm, s);
  }
  red[t] = lm; __syncthreads();
  for (int o = 32; o > 0; o >>= 1) { if (t < o) red[t] = fmaxf(red[t], red[t + o]); __syncthreads(); }
  const float mx = red[0]; __syncthreads();
  float sm = 0.0f;
#pragma unroll
  for (int jj = 0; jj < 4; ++jj) {
    const int j = t * 4 + jj;
    const float e = expf(p[j] - mx); p[j] = e; sm += e;
  }
  red[t] = sm; __syncthreads();
  for (int o = 32; o > 0; o >>= 1) { if (t < o) red[t] += red[t + o]; __syncthreads(); }
  const float inv = 1.0f / red[0]; __syncthreads();
  float acc = 0.0f;
  for (int j = 0; j < MLAND; ++j) acc += p[j] * WZ[((size_t)h * MLAND + j) * DH + t];
  acc *= inv;
  // depthwise residual conv along sequence (pad 16 each side)
  const float* w = resw + h * 33;
  float cv = 0.0f;
  for (int tap = 0; tap < 33; ++tap) {
    const int nn = n + tap - 16;
    if (nn >= 0 && nn < NPAD) cv += w[tap] * Vb[((size_t)h * NPAD + nn) * DH + t];
  }
  OUT[(size_t)n * DMODEL + h * DH + t] = acc + cv;
}

// PPEG: y = f + conv7(f)+b7 + conv5(f)+b5 + conv3(f)+b3 over 100x100 grid,
// written back into X tokens 1..NTOK. F is a snapshot copy (token-major).
__global__ void k_ppeg(const float* __restrict__ F,
                       const float* __restrict__ w7, const float* __restrict__ b7,
                       const float* __restrict__ w5, const float* __restrict__ b5,
                       const float* __restrict__ w3, const float* __restrict__ b3,
                       float* __restrict__ X) {
  const int idx = blockIdx.x * blockDim.x + threadIdx.x;
  if (idx >= NTOK * DMODEL) return;
  const int c = idx & 511;
  const int n = idx >> 9;
  const int hh = n / HGRID, ww = n % HGRID;
  float acc = F[idx];                    // identity term
  const float* k7 = w7 + (size_t)c * 49;
  for (int dy = -3; dy <= 3; ++dy)
    for (int dx = -3; dx <= 3; ++dx) {
      const int yy = hh + dy, xx = ww + dx;
      if (yy >= 0 && yy < HGRID && xx >= 0 && xx < HGRID)
        acc += k7[(dy + 3) * 7 + (dx + 3)] * F[(size_t)(yy * HGRID + xx) * DMODEL + c];
    }
  const float* k5 = w5 + (size_t)c * 25;
  for (int dy = -2; dy <= 2; ++dy)
    for (int dx = -2; dx <= 2; ++dx) {
      const int yy = hh + dy, xx = ww + dx;
      if (yy >= 0 && yy < HGRID && xx >= 0 && xx < HGRID)
        acc += k5[(dy + 2) * 5 + (dx + 2)] * F[(size_t)(yy * HGRID + xx) * DMODEL + c];
    }
  const float* k3 = w3 + (size_t)c * 9;
  for (int dy = -1; dy <= 1; ++dy)
    for (int dx = -1; dx <= 1; ++dx) {
      const int yy = hh + dy, xx = ww + dx;
      if (yy >= 0 && yy < HGRID && xx >= 0 && xx < HGRID)
        acc += k3[(dy + 1) * 3 + (dx + 1)] * F[(size_t)(yy * HGRID + xx) * DMODEL + c];
    }
  X[(size_t)(n + 1) * DMODEL + c] = acc + b7[c] + b5[c] + b3[c];
}

// ---------------- pooling head ----------------------------------------------
__global__ void k_head_vec(const float* __restrict__ AH, const float* __restrict__ Wa2,
                           const float* __restrict__ ba2, float* __restrict__ AVEC,
                           float* __restrict__ out_araw) {
  const int i = blockIdx.x * blockDim.x + threadIdx.x;
  if (i >= NTOK) return;
  const float* r = AH + (size_t)i * AHID;
  float s = 0.0f;
  for (int d = 0; d < AHID; ++d) s += r[d] * Wa2[d];
  s += ba2[0];
  AVEC[i] = s;
  out_araw[i] = s;                       // A_raw (pre-softmax)
}

__global__ void k_softmax_stats(const float* __restrict__ AVEC, float* __restrict__ SCAL) {
  __shared__ float red[256];
  const int t = threadIdx.x;
  float mx = -3.4e38f;
  for (int i = t; i < NTOK; i += 256) mx = fmaxf(mx, AVEC[i]);
  red[t] = mx; __syncthreads();
  for (int o = 128; o > 0; o >>= 1) { if (t < o) red[t] = fmaxf(red[t], red[t + o]); __syncthreads(); }
  mx = red[0]; __syncthreads();
  float sm = 0.0f;
  for (int i = t; i < NTOK; i += 256) sm += expf(AVEC[i] - mx);
  red[t] = sm; __syncthreads();
  for (int o = 128; o > 0; o >>= 1) { if (t < o) red[t] += red[t + o]; __syncthreads(); }
  if (t == 0) { SCAL[2] = mx; SCAL[3] = red[0]; }
}

__global__ void k_attnpool(const float* __restrict__ AVEC, const float* __restrict__ SCAL,
                           const float* __restrict__ X, float* __restrict__ MV) {
  const int d = blockIdx.x;             // 512
  const int t = threadIdx.x;            // 256
  __shared__ float red[256];
  const float mx = SCAL[2];
  const float inv = 1.0f / SCAL[3];
  float s = 0.0f;
  for (int i = t; i < NTOK; i += 256)
    s += expf(AVEC[i] - mx) * X[(size_t)(i + 1) * DMODEL + d];
  red[t] = s; __syncthreads();
  for (int o = 128; o > 0; o >>= 1) { if (t < o) red[t] += red[t + o]; __syncthreads(); }
  if (t == 0) MV[d] = red[0] * inv;
}

__global__ void k_logits(const float* __restrict__ MV, const float* __restrict__ Wc,
                         const float* __restrict__ bc, float* __restrict__ out) {
  __shared__ float l[2];
  const int t = threadIdx.x;
  if (t < 2) {
    float s = 0.0f;
    for (int d = 0; d < DMODEL; ++d) s += MV[d] * Wc[d * 2 + t];
    l[t] = s + bc[t];
  }
  __syncthreads();
  if (t == 0) {
    const float l0 = l[0], l1 = l[1];
    out[0] = l0; out[1] = l1;
    const float m = fmaxf(l0, l1);
    const float e0 = expf(l0 - m), e1 = expf(l1 - m), ss = e0 + e1;
    out[2] = e0 / ss; out[3] = e1 / ss;
    out[4] = (l1 > l0) ? 1.0f : 0.0f;   // argmax (first max wins tie)
  }
}

// ---------------- host orchestration ----------------------------------------
static inline int gemm_blocks(int M, int N) {
  const int tiles = ((M + 15) / 16) * (N / 64);   // 16x64 strip per wave
  return (tiles + 3) / 4;                          // 4 waves per 128-thread block
}
static inline dim3 bmm_grid(int Md, int Nd) {
  const int waves = (Md / 16) * (Nd / 16);
  return dim3((waves + 3) / 4, 1, HEADS);
}

static void attention_block(hipStream_t stream, float* X, float* XLN,
                            float* Qb, float* Kb, float* Vb, float* QL, float* KL,
                            float* A2, float* Zb, float* ZN, float* XZ, float* TA,
                            float* TB, float* W3, float* WZ, float* SCAL,
                            const float* ln_g, const float* ln_b, const float* Wqkv,
                            const float* Wo, const float* bo, const float* resw) {
  k_ln_pad<<<NPAD, 256, 0, stream>>>(X, ln_g, ln_b, XLN);
  k_wmma_qkv<<<gemm_blocks(NPAD, 3 * DMODEL), 128, 0, stream>>>(XLN, Wqkv, Qb, Kb, Vb);
  k_landmarks<<<HEADS * MLAND, DH, 0, stream>>>(Qb, QL);
  k_landmarks<<<HEADS * MLAND, DH, 0, stream>>>(Kb, KL);
  k_a2<<<HEADS * MLAND, MLAND, 0, stream>>>(QL, KL, A2);
  k_zero4<<<1, 8, 0, stream>>>(SCAL);
  {
    dim3 g((HEADS * MLAND + 255) / 256, 2);
    k_rowcol<<<g, 256, 0, stream>>>(A2, SCAL);
  }
  const int nel = HEADS * MLAND * MLAND;
  k_pinv_init<<<(nel + 255) / 256, 256, 0, stream>>>(A2, SCAL, Zb);
  float* Zcur = Zb;
  float* Znxt = ZN;
  for (int it = 0; it < PINVIT; ++it) {
    k_bmm<<<bmm_grid(MLAND, MLAND), 128, 0, stream>>>(A2, Zcur, XZ, MLAND, MLAND, MLAND, 1.0f);
    k_diagsub<<<(nel + 255) / 256, 256, 0, stream>>>(XZ, TA, 7.0f);
    k_bmm<<<bmm_grid(MLAND, MLAND), 128, 0, stream>>>(XZ, TA, TB, MLAND, MLAND, MLAND, 1.0f);
    k_diagsub<<<(nel + 255) / 256, 256, 0, stream>>>(TB, TA, 15.0f);
    k_bmm<<<bmm_grid(MLAND, MLAND), 128, 0, stream>>>(XZ, TA, TB, MLAND, MLAND, MLAND, 1.0f);
    k_diagsub<<<(nel + 255) / 256, 256, 0, stream>>>(TB, TA, 13.0f);
    k_bmm<<<bmm_grid(MLAND, MLAND), 128, 0, stream>>>(Zcur, TA, Znxt, MLAND, MLAND, MLAND, 0.25f);
    float* tmp = Zcur; Zcur = Znxt; Znxt = tmp;
  }
  k_a3v<<<HEADS * MLAND, 256, 0, stream>>>(QL, Kb, Vb, W3);
  k_bmm<<<bmm_grid(MLAND, DH), 128, 0, stream>>>(Zcur, W3, WZ, MLAND, DH, MLAND, 1.0f);
  {
    dim3 g(NPAD, HEADS);
    k_attn_out<<<g, 64, 0, stream>>>(Qb, KL, WZ, Vb, resw, XLN);  // XLN reused as OUT
  }
  // residual: X += (OUT rows PADROWS.. ) @ Wo + bo   (last TTOK rows)
  k_wmma_gemm<0, 1><<<gemm_blocks(TTOK, DMODEL), 128, 0, stream>>>(
      XLN + (size_t)PADROWS * DMODEL, Wo, bo, X, TTOK, DMODEL, DMODEL);
}

extern "C" void kernel_launch(void* const* d_in, const int* in_sizes, int n_in,
                              void* d_out, int out_size, void* d_ws, size_t ws_size,
                              hipStream_t stream) {
  const float* h      = (const float*)d_in[0];
  const float* W_fc1  = (const float*)d_in[1];
  const float* b_fc1  = (const float*)d_in[2];
  const float* cls    = (const float*)d_in[3];
  const float* ln1_g  = (const float*)d_in[4];
  const float* ln1_b  = (const float*)d_in[5];
  const float* qkv1   = (const float*)d_in[6];
  const float* out1_w = (const float*)d_in[7];
  const float* out1_b = (const float*)d_in[8];
  const float* res1_w = (const float*)d_in[9];
  const float* w7     = (const float*)d_in[10];
  const float* b7     = (const float*)d_in[11];
  const float* w5     = (const float*)d_in[12];
  const float* b5     = (const float*)d_in[13];
  const float* w3     = (const float*)d_in[14];
  const float* b3     = (const float*)d_in[15];
  const float* ln2_g  = (const float*)d_in[16];
  const float* ln2_b  = (const float*)d_in[17];
  const float* qkv2   = (const float*)d_in[18];
  const float* out2_w = (const float*)d_in[19];
  const float* out2_b = (const float*)d_in[20];
  const float* res2_w = (const float*)d_in[21];
  const float* Wa1    = (const float*)d_in[22];
  const float* ba1    = (const float*)d_in[23];
  const float* Wa2    = (const float*)d_in[24];
  const float* ba2    = (const float*)d_in[25];
  const float* Wc     = (const float*)d_in[26];
  const float* bc     = (const float*)d_in[27];
  float* out = (float*)d_out;

  float* ws = (float*)d_ws;
  size_t off = 0;
  auto alloc = [&](size_t n) { float* p = ws + off; off += n; return p; };
  float* X    = alloc((size_t)TTOK * DMODEL);
  float* XLN  = alloc((size_t)NPAD * DMODEL);   // doubles as attention OUT
  float* Qb   = alloc((size_t)NPAD * DMODEL);
  float* Kb   = alloc((size_t)NPAD * DMODEL);
  float* Vb   = alloc((size_t)NPAD * DMODEL);
  float* QL   = alloc((size_t)HEADS * MLAND * DH);
  float* KL   = alloc((size_t)HEADS * MLAND * DH);
  float* A2   = alloc((size_t)HEADS * MLAND * MLAND);
  float* Zb   = alloc((size_t)HEADS * MLAND * MLAND);
  float* ZN   = alloc((size_t)HEADS * MLAND * MLAND);
  float* XZ   = alloc((size_t)HEADS * MLAND * MLAND);
  float* TA   = alloc((size_t)HEADS * MLAND * MLAND);
  float* TB   = alloc((size_t)HEADS * MLAND * MLAND);
  float* W3b  = alloc((size_t)HEADS * MLAND * DH);
  float* WZ   = alloc((size_t)HEADS * MLAND * DH);
  float* AH   = alloc((size_t)NTOK * AHID);
  float* AVEC = alloc((size_t)NTOK);
  float* MV   = alloc((size_t)DMODEL);
  float* SCAL = alloc(8);
  (void)in_sizes; (void)n_in; (void)out_size; (void)ws_size;

  // x = relu(h @ W_fc1 + b); rows 1..NTOK of X; row 0 = cls token.
  k_wmma_gemm<1, 0><<<gemm_blocks(NTOK, DMODEL), 128, 0, stream>>>(
      h, W_fc1, b_fc1, X + DMODEL, NTOK, DMODEL, EIN);
  k_set_cls<<<1, DMODEL, 0, stream>>>(cls, X);

  attention_block(stream, X, XLN, Qb, Kb, Vb, QL, KL, A2, Zb, ZN, XZ, TA, TB,
                  W3b, WZ, SCAL, ln1_g, ln1_b, qkv1, out1_w, out1_b, res1_w);

  // PPEG (snapshot feature map into Qb, free at this point)
  hipMemcpyAsync(Qb, X + DMODEL, (size_t)NTOK * DMODEL * sizeof(float),
                 hipMemcpyDeviceToDevice, stream);
  k_ppeg<<<(NTOK * DMODEL + 255) / 256, 256, 0, stream>>>(Qb, w7, b7, w5, b5, w3, b3, X);

  attention_block(stream, X, XLN, Qb, Kb, Vb, QL, KL, A2, Zb, ZN, XZ, TA, TB,
                  W3b, WZ, SCAL, ln2_g, ln2_b, qkv2, out2_w, out2_b, res2_w);

  // pooling head: AH = tanh(pf @ Wa1 + ba1)
  k_wmma_gemm<2, 0><<<gemm_blocks(NTOK, AHID), 128, 0, stream>>>(
      X + DMODEL, Wa1, ba1, AH, NTOK, AHID, DMODEL);
  k_head_vec<<<(NTOK + 255) / 256, 256, 0, stream>>>(AH, Wa2, ba2, AVEC, out + 5);
  k_softmax_stats<<<1, 256, 0, stream>>>(AVEC, SCAL);
  k_attnpool<<<DMODEL, 256, 0, stream>>>(AVEC, SCAL, X, MV);
  k_logits<<<1, 64, 0, stream>>>(MV, Wc, bc, out);
}
